// GCCF_80960133529714
// MI455X (gfx1250) — matrix operand order repaired and checked by reading, not compiled, not run on hardware
//
#include <hip/hip_runtime.h>

#define GNU 16000
#define GNM 8000
#define GE  64
#define GB  100000

typedef __attribute__((ext_vector_type(16))) __bf16 bf16x16;
typedef __attribute__((ext_vector_type(8)))  __bf16 bf16x8;
typedef __attribute__((ext_vector_type(2)))  __bf16 bf16x2;
typedef __attribute__((ext_vector_type(8)))  float  f32x8;

#define BSTRIDE 40   // padded k-stride (halves) for neighbor LDS tile: 80B rows (16B aligned)
#define XSTRIDE 72   // padded k-stride (halves) for per-wave x tile: 144B rows (16B aligned)

#define WMMA_BF16(A, B, C) \
  __builtin_amdgcn_wmma_f32_16x16x32_bf16(false, (A), false, (B), (short)0, (C), false, false)

__device__ __forceinline__ bf16x16 pack16(float4 a, float4 b, float4 c, float4 d) {
  bf16x16 r;
  r[0]=(__bf16)a.x;  r[1]=(__bf16)a.y;  r[2]=(__bf16)a.z;  r[3]=(__bf16)a.w;
  r[4]=(__bf16)b.x;  r[5]=(__bf16)b.y;  r[6]=(__bf16)b.z;  r[7]=(__bf16)b.w;
  r[8]=(__bf16)c.x;  r[9]=(__bf16)c.y;  r[10]=(__bf16)c.z; r[11]=(__bf16)c.w;
  r[12]=(__bf16)d.x; r[13]=(__bf16)d.y; r[14]=(__bf16)d.z; r[15]=(__bf16)d.w;
  return r;
}

__device__ __forceinline__ bf16x16 cat8(bf16x8 lo, bf16x8 hi) {
  bf16x16 r;
  #pragma unroll
  for (int i = 0; i < 8; ++i) { r[i] = lo[i]; r[8 + i] = hi[i]; }
  return r;
}

// Stage a [32 x 64] f32 chunk of `nbr` into LDS as bf16, [n][k] layout, k-pairs packed.
// Lanes map to consecutive n -> fully coalesced 256B global reads per 16 lanes.
__device__ __forceinline__ void stage_chunk(const float* __restrict__ nbr, int k0,
                                            __bf16* __restrict__ buf, int tid) {
  const int kp = (tid >> 4) << 1;        // even k: 0,2,...,30
  const int n4 = (tid & 15) << 2;        // 0,4,...,60
  const float4 va = *(const float4*)(nbr + (size_t)(k0 + kp) * GE + n4);
  const float4 vb = *(const float4*)(nbr + (size_t)(k0 + kp + 1) * GE + n4);
  unsigned int* bp = (unsigned int*)buf; // dword view; row stride = BSTRIDE/2 dwords
  const int kd = kp >> 1;
  bf16x2 p0; p0[0] = (__bf16)va.x; p0[1] = (__bf16)vb.x;
  bf16x2 p1; p1[0] = (__bf16)va.y; p1[1] = (__bf16)vb.y;
  bf16x2 p2; p2[0] = (__bf16)va.z; p2[1] = (__bf16)vb.z;
  bf16x2 p3; p3[0] = (__bf16)va.w; p3[1] = (__bf16)vb.w;
  bp[(n4 + 0) * (BSTRIDE / 2) + kd] = __builtin_bit_cast(unsigned int, p0);
  bp[(n4 + 1) * (BSTRIDE / 2) + kd] = __builtin_bit_cast(unsigned int, p1);
  bp[(n4 + 2) * (BSTRIDE / 2) + kd] = __builtin_bit_cast(unsigned int, p2);
  bp[(n4 + 3) * (BSTRIDE / 2) + kd] = __builtin_bit_cast(unsigned int, p3);
}

// One GCN layer half:  out = leaky_relu((adj @ nbr + self) @ Wgt^T + 2*bias)
// Each wave owns a 32-row (two 16-row M-tiles) x 64-col output slab.
__global__ __launch_bounds__(256)
void gcn_layer(const float* __restrict__ adj,   // [rows x K]
               const float* __restrict__ nbr,   // [K x 64]
               const float* __restrict__ self,  // [rows x 64]
               const float* __restrict__ Wgt,   // [64 x 64]
               const float* __restrict__ bias,  // [64]
               float* __restrict__ outp,        // [rows x 64]
               int rows, int K)
{
  __shared__ __bf16 Bsh[2][64 * BSTRIDE];       // double-buffered neighbor chunk
  __shared__ __bf16 Xsh[8 * 32 * XSTRIDE];      // per-wave 32-row x tile

  const int tid  = threadIdx.x;
  const int wave = tid >> 5;
  const int lane = tid & 31;
  const int half = lane >> 4;                   // 0: lanes 0-15, 1: lanes 16-31
  const int l16  = lane & 15;

  const int ntp = rows >> 5;                    // 32-row tile pairs
  int tp = blockIdx.x * 8 + wave;
  const bool active = (tp < ntp);
  if (!active) tp = ntp - 1;                    // clamp: uniform barrier participation
  const int rowBase = tp << 5;

  f32x8 acc[2][4] = {};

  const float* arow0 = adj + (size_t)(rowBase + l16)      * (size_t)K;
  const float* arow1 = adj + (size_t)(rowBase + 16 + l16) * (size_t)K;
  const int koff = half ? 8 : 0;                // A layout K sub-offset per lane half
  const int brow = half ? 8 : 0;                // B layout K base per lane half

  const int nchunks = K >> 5;
  stage_chunk(nbr, 0, Bsh[0], tid);             // prologue: chunk 0 -> buf 0

  for (int ic = 0; ic < nchunks; ++ic) {
    const int k0 = ic << 5;
    __syncthreads();                            // buf[ic&1] staged; buf[(ic+1)&1] free
    if (ic + 1 < nchunks)
      stage_chunk(nbr, k0 + 32, Bsh[(ic + 1) & 1], tid);

    // prefetch adjacency stream two chunks ahead (HBM-bound stream)
    __builtin_prefetch(arow0 + k0 + 64, 0, 1);
    __builtin_prefetch(arow1 + k0 + 64, 0, 1);

    // ---- A fragments for both M-tiles (f32 -> bf16 in-register) ----
    const int kb = k0 + koff;
    const bf16x16 af0 = pack16(*(const float4*)(arow0 + kb),
                               *(const float4*)(arow0 + kb + 4),
                               *(const float4*)(arow0 + kb + 16),
                               *(const float4*)(arow0 + kb + 20));
    const bf16x16 af1 = pack16(*(const float4*)(arow1 + kb),
                               *(const float4*)(arow1 + kb + 4),
                               *(const float4*)(arow1 + kb + 16),
                               *(const float4*)(arow1 + kb + 20));

    // ---- all 4 B fragments up front (lets waits be graduated) ----
    const __bf16* bbuf = Bsh[ic & 1];
    bf16x8 blo[4], bhi[4];
    #pragma unroll
    for (int t = 0; t < 4; ++t) {
      const __bf16* bp = bbuf + (t * 16 + l16) * BSTRIDE + brow;
      blo[t] = *(const bf16x8*)(bp);
      bhi[t] = *(const bf16x8*)(bp + 16);
    }

    // ---- 8 WMMAs: both M-tiles share each B fragment ----
    #pragma unroll
    for (int t = 0; t < 4; ++t) {
      const bf16x16 bf = cat8(blo[t], bhi[t]);
      acc[0][t] = WMMA_BF16(af0, bf, acc[0][t]);
      acc[1][t] = WMMA_BF16(af1, bf, acc[1][t]);
    }
  }

  // ---- epilogue: x = acc + self; stage 32x64 tile to LDS in A layout ----
  __bf16* X = Xsh + wave * 32 * XSTRIDE;
  #pragma unroll
  for (int s = 0; s < 2; ++s) {
    #pragma unroll
    for (int t = 0; t < 4; ++t) {
      #pragma unroll
      for (int v = 0; v < 8; ++v) {
        const int M = v + (half ? 8 : 0);       // C layout: VGPR v -> row M
        const int N = t * 16 + l16;
        const float x = acc[s][t][v] + self[(size_t)(rowBase + s * 16 + M) * GE + N];
        X[(s * 16 + M) * XSTRIDE + N] = (__bf16)x;
      }
    }
  }
  __syncthreads();

  // ---- y = x @ Wgt^T : per 16-row subtile, 2 k-chunks x 4 n-tiles of WMMA ----
  #pragma unroll
  for (int s = 0; s < 2; ++s) {
    f32x8 o[4] = {};
    #pragma unroll
    for (int kc = 0; kc < 2; ++kc) {
      const int kb2 = kc * 32 + koff;
      const __bf16* xp = X + (s * 16 + l16) * XSTRIDE;
      const bf16x16 af = cat8(*(const bf16x8*)(xp + kb2),
                              *(const bf16x8*)(xp + kb2 + 16));
      #pragma unroll
      for (int t = 0; t < 4; ++t) {
        // B[k][n] = Wgt[n][k]; lane supplies n = t*16+l16, k runs 8h..8h+7 and 16+8h..16+8h+7
        const float* wp = Wgt + (t * 16 + l16) * GE + kc * 32 + (half ? 8 : 0);
        o[t] = WMMA_BF16(af, pack16(*(const float4*)(wp),
                                    *(const float4*)(wp + 4),
                                    *(const float4*)(wp + 16),
                                    *(const float4*)(wp + 20)), o[t]);
      }
    }
    if (active) {
      #pragma unroll
      for (int t = 0; t < 4; ++t) {
        #pragma unroll
        for (int v = 0; v < 8; ++v) {
          const int M = v + (half ? 8 : 0);
          const int N = t * 16 + l16;
          float val = o[t][v] + 2.0f * bias[N];
          val = fmaxf(val, 0.01f * val);        // leaky_relu(0.01)
          outp[(size_t)(rowBase + s * 16 + M) * GE + N] = val;
        }
      }
    }
  }
}

// out[b] = bo + sum_l sum_e U_l[uid[b]][e] * M_l[mid[b]][e] * Wo[l*64+e]
__global__ __launch_bounds__(256)
void gather_dot(const float* __restrict__ u0, const float* __restrict__ u1,
                const float* __restrict__ u2, const float* __restrict__ u3,
                const float* __restrict__ m0, const float* __restrict__ m1,
                const float* __restrict__ m2, const float* __restrict__ m3,
                const float* __restrict__ Wo, const float* __restrict__ bo,
                const int* __restrict__ uid, const int* __restrict__ mid,
                float* __restrict__ out, int B)
{
  const int b = blockIdx.x * 256 + threadIdx.x;
  if (b >= B) return;
  const float* us[4] = {u0, u1, u2, u3};
  const float* ms[4] = {m0, m1, m2, m3};
  const size_t ur = (size_t)uid[b] * GE;
  const size_t mr = (size_t)mid[b] * GE;
  float s = bo[0];
  #pragma unroll
  for (int l = 0; l < 4; ++l) {
    const float* up = us[l] + ur;
    const float* mp = ms[l] + mr;
    const float* wp = Wo + l * GE;
    #pragma unroll
    for (int e = 0; e < GE; e += 4) {
      const float4 uu = *(const float4*)(up + e);
      const float4 mm = *(const float4*)(mp + e);
      const float4 ww = *(const float4*)(wp + e);
      s += uu.x * mm.x * ww.x + uu.y * mm.y * ww.y
         + uu.z * mm.z * ww.z + uu.w * mm.w * ww.w;
    }
  }
  out[b] = s;
}

extern "C" void kernel_launch(void* const* d_in, const int* in_sizes, int n_in,
                              void* d_out, int out_size, void* d_ws, size_t ws_size,
                              hipStream_t stream)
{
  const float* user_adj  = (const float*)d_in[0];
  const float* movie_adj = (const float*)d_in[1];
  const float* user_emb  = (const float*)d_in[2];
  const float* movie_emb = (const float*)d_in[3];
  const float* Wu        = (const float*)d_in[4];
  const float* bu        = (const float*)d_in[5];
  const float* Wm        = (const float*)d_in[6];
  const float* bm        = (const float*)d_in[7];
  const float* Wo        = (const float*)d_in[8];
  const float* bo        = (const float*)d_in[9];
  const int*   uid       = (const int*)d_in[10];
  const int*   mid       = (const int*)d_in[11];

  // workspace: u1,u2,u3 [NU*64] then m1,m2,m3 [NM*64] (f32) -> ~17.6 MB
  float* ws = (float*)d_ws;
  float* ubuf[3] = { ws, ws + (size_t)GNU * GE, ws + (size_t)2 * GNU * GE };
  float* mbuf[3] = { ws + (size_t)3 * GNU * GE,
                     ws + (size_t)3 * GNU * GE + (size_t)GNM * GE,
                     ws + (size_t)3 * GNU * GE + (size_t)2 * GNM * GE };

  const float* uprev = user_emb;
  const float* mprev = movie_emb;
  for (int l = 0; l < 3; ++l) {
    // u_{l+1}: 16000 rows -> 500 tile-pairs / 8 waves -> 63 blocks
    gcn_layer<<<63, 256, 0, stream>>>(user_adj, mprev, uprev,
                                      Wu + (size_t)l * GE * GE, bu + l * GE,
                                      ubuf[l], GNU, GNM);
    // m_{l+1}: 8000 rows -> 250 tile-pairs -> 32 blocks
    gcn_layer<<<32, 256, 0, stream>>>(movie_adj, uprev, mprev,
                                      Wm + (size_t)l * GE * GE, bm + l * GE,
                                      mbuf[l], GNM, GNU);
    uprev = ubuf[l];
    mprev = mbuf[l];
  }

  gather_dot<<<(GB + 255) / 256, 256, 0, stream>>>(
      user_emb, ubuf[0], ubuf[1], ubuf[2],
      movie_emb, mbuf[0], mbuf[1], mbuf[2],
      Wo, bo, uid, mid, (float*)d_out, GB);
}
